// SelfAttentionFusion_63075889709441
// MI455X (gfx1250) — matrix operand order repaired and checked by reading, not compiled, not run on hardware
//
#include <hip/hip_runtime.h>

typedef __attribute__((ext_vector_type(16))) __bf16 v16bf;
typedef __attribute__((ext_vector_type(8)))  float  v8f;

// ---------------------------------------------------------------------------
// WMMA fragment loaders (wave32, v_wmma_f32_16x16x32_bf16; A/B in contiguous k order, not the ISA table order)
// A (16x32 bf16): lane = half*16 + (M%16); per-lane pairs at
//   k = half*16 + v*2  (v = vgpr index 0..7)
// B (32x16 bf16): lane covers column N = lane&15; half-wave covers K block of 16.
//   Loading rows of W (N x K) row-major gives B = W^T directly (X @ W^T shape).
// C/D (16x16 f32): row = half*8 + r, col = lane&15.
// ---------------------------------------------------------------------------
__device__ __forceinline__ v16bf frag_a(const __bf16* base, int row0, int ldk,
                                        int k0, int lane) {
  const int row  = row0 + (lane & 15);
  const int half = lane >> 4;
  const __bf16* p = base + row * ldk + k0;
  union { v16bf v; unsigned int u[8]; } r;
#pragma unroll
  for (int vv = 0; vv < 8; ++vv) {
    const int k = (half << 4) + (vv << 1);
    r.u[vv] = *(const unsigned int*)(p + k);
  }
  return r.v;
}

__device__ __forceinline__ v16bf frag_b(const __bf16* base, int col0, int ldk,
                                        int k0, int lane) {
  const int n    = col0 + (lane & 15);
  const int half = lane >> 4;
  const __bf16* p = base + n * ldk + k0 + half * 16;
  return *(const v16bf*)p;  // 16 contiguous K values
}

// B fragment where B (K x N) is stored row-major with row stride `stride`
// (used for attn @ V: columns of V).
__device__ __forceinline__ v16bf frag_b_strided(const __bf16* base, int col0,
                                                int stride, int k0, int lane) {
  const int n    = col0 + (lane & 15);
  const int half = lane >> 4;
  const __bf16* p = base + n + (k0 + half * 16) * stride;
  v16bf r;
#pragma unroll
  for (int e = 0; e < 16; ++e) r[e] = p[e * stride];
  return r;
}

// ---------------------------------------------------------------------------
// Kernel 1: one-time fp32 -> bf16 conversion of all weight matrices into ws.
// ws layout (bf16 elements): conv_w @0 (4096), trans_w @4096 (4096),
// qkv_w @8192 (49152), proj_w @57344 (16384). Total 73728 elements.
// ---------------------------------------------------------------------------
__global__ void convert_weights(const float* __restrict__ cw,
                                const float* __restrict__ tw,
                                const float* __restrict__ qw,
                                const float* __restrict__ pw,
                                unsigned short* __restrict__ out) {
  const int i = blockIdx.x * blockDim.x + threadIdx.x;
  if (i >= 73728) return;
  float f;
  if (i < 4096)       f = cw[i];
  else if (i < 8192)  f = tw[i - 4096];
  else if (i < 57344) f = qw[i - 8192];
  else                f = pw[i - 57344];
  union { __bf16 h; unsigned short u; } cvt;
  cvt.h = (__bf16)f;
  out[i] = cvt.u;
}

// ---------------------------------------------------------------------------
// Kernel 2: fully fused per-window pipeline. One block (4 waves) per 4x4x4
// window; N=64 tokens, DIM=128, 4 heads x 32.
// Dynamic LDS (160 KB):
//   [0      ,16384) sX (64x128 bf16)      -> reused as sO
//   [16384  ,32768) sQ (64x128 bf16, pre-scaled)
//   [32768  ,49152) sK
//   [49152  ,65536) sV
//   [65536  ,131072) sAtt (4x64x64 f32)   -> stage1 sXcb/sXtb live here,
//                                            later reused as sP (64x128 f32)
//   [131072 ,163840) sAttB (4x64x64 bf16)
// ---------------------------------------------------------------------------
__global__ void __launch_bounds__(128)
fused_window_attn(const float* __restrict__ conv_feat,
                  const float* __restrict__ trans_feat,
                  const float* __restrict__ conv_b,
                  const float* __restrict__ trans_b,
                  const float* __restrict__ qkv_b,
                  const float* __restrict__ proj_b,
                  const float* __restrict__ rel_table,
                  const int*   __restrict__ rel_index,
                  const unsigned short* __restrict__ wbf_u,
                  float* __restrict__ out) {
  extern __shared__ unsigned char smem[];
  __bf16* sX    = (__bf16*)(smem);
  __bf16* sQ    = (__bf16*)(smem + 16384);
  __bf16* sK    = (__bf16*)(smem + 32768);
  __bf16* sV    = (__bf16*)(smem + 49152);
  float*  sAtt  = (float*) (smem + 65536);
  __bf16* sXcb  = (__bf16*)(smem + 65536);   // stage-1 only (8 KB)
  __bf16* sXtb  = (__bf16*)(smem + 73728);   // stage-1 only (8 KB)
  __bf16* sAttB = (__bf16*)(smem + 131072);

  const __bf16* wbf = (const __bf16*)wbf_u;

  const int tid  = threadIdx.x;
  const int wave = tid >> 5;
  const int lane = tid & 31;

  const int wb = blockIdx.x;                  // 16x16x16 windows
  const int wd = wb >> 8, wh = (wb >> 4) & 15, ww = wb & 15;

  // ---- Stage 1: gather window voxels, fp32 -> bf16 (token-major [n][c]) ----
  for (int i = tid; i < 4096; i += 128) {
    const int c = i >> 6, n = i & 63;
    const int gd = wd * 4 + (n >> 4);
    const int gh = wh * 4 + ((n >> 2) & 3);
    const int gw = ww * 4 + (n & 3);
    const size_t g = (((size_t)c * 64 + gd) * 64 + gh) * 64 + gw;
    sXcb[n * 64 + c] = (__bf16)conv_feat[g];
    sXtb[n * 64 + c] = (__bf16)trans_feat[g];
  }
  __syncthreads();

  // ---- Stage 2: channel projections (two 64x64x64 GEMMs) -> sX 64x128 -----
  for (int br = 0; br < 2; ++br) {
    const __bf16* A    = br ? sXtb : sXcb;
    const __bf16* B    = wbf + (br ? 4096 : 0);
    const float*  bias = br ? trans_b : conv_b;
    for (int t = wave; t < 16; t += 4) {
      const int tm = t & 3, tn = t >> 2;
      v8f acc = {};
#pragma unroll
      for (int kk = 0; kk < 2; ++kk) {
        v16bf a = frag_a(A, tm * 16, 64, kk * 32, lane);
        v16bf b = frag_b(B, tn * 16, 64, kk * 32, lane);
        acc = __builtin_amdgcn_wmma_f32_16x16x32_bf16(false, a, false, b,
                                                      (short)0, acc, false, false);
      }
      const int col  = tn * 16 + (lane & 15);
      const int half = lane >> 4;
      const float bv = bias[col];
#pragma unroll
      for (int r = 0; r < 8; ++r)
        sX[(tm * 16 + half * 8 + r) * 128 + br * 64 + col] = (__bf16)(acc[r] + bv);
    }
  }
  __syncthreads();

  // ---- Stage 3: QKV projections (64x128 @ 128x128 each) -------------------
  const float qscale = 0.17677669529663687f;  // 1/sqrt(32)
  for (int j = 0; j < 3; ++j) {
    __bf16* dst = (j == 0) ? sQ : (j == 1) ? sK : sV;
    const __bf16* B = wbf + 8192 + j * 16384;  // rows of Wq/Wk/Wv (N x K)
    for (int t = wave; t < 32; t += 4) {
      const int tm = t & 3, tn = t >> 2;
      v8f acc = {};
#pragma unroll
      for (int kk = 0; kk < 4; ++kk) {
        v16bf a = frag_a(sX, tm * 16, 128, kk * 32, lane);
        v16bf b = frag_b(B, tn * 16, 128, kk * 32, lane);
        acc = __builtin_amdgcn_wmma_f32_16x16x32_bf16(false, a, false, b,
                                                      (short)0, acc, false, false);
      }
      const int col  = tn * 16 + (lane & 15);
      const int half = lane >> 4;
      const float bv = qkv_b[j * 128 + col];
#pragma unroll
      for (int r = 0; r < 8; ++r) {
        float v = acc[r] + bv;
        if (j == 0) v *= qscale;
        dst[(tm * 16 + half * 8 + r) * 128 + col] = (__bf16)v;
      }
    }
  }
  __syncthreads();

  // ---- Stage 4: logits Q @ K^T per head (K=32, one WMMA/tile) + bias ------
  for (int t = wave; t < 64; t += 4) {
    const int h = t >> 4, rem = t & 15, tm = rem & 3, tn = rem >> 2;
    v16bf a = frag_a(sQ + h * 32, tm * 16, 128, 0, lane);
    v16bf b = frag_b(sK + h * 32, tn * 16, 128, 0, lane);  // rows of K = B
    v8f acc = {};
    acc = __builtin_amdgcn_wmma_f32_16x16x32_bf16(false, a, false, b,
                                                  (short)0, acc, false, false);
    const int m    = tn * 16 + (lane & 15);
    const int half = lane >> 4;
#pragma unroll
    for (int r = 0; r < 8; ++r) {
      const int n   = tm * 16 + half * 8 + r;
      const int idx = rel_index[n * 64 + m];
      sAtt[(h * 64 + n) * 64 + m] = acc[r] + rel_table[idx * 4 + h];
    }
  }
  __syncthreads();

  // ---- Stage 5: softmax over keys (256 rows, 2 per thread) ----------------
  for (int row = tid; row < 256; row += 128) {
    float* p = sAtt + row * 64;
    float mx = -3.0e38f;
    for (int m = 0; m < 64; ++m) mx = fmaxf(mx, p[m]);
    float s = 0.f;
    for (int m = 0; m < 64; ++m) s += __expf(p[m] - mx);
    const float rs = 1.f / s;
    __bf16* q = sAttB + row * 64;
    for (int m = 0; m < 64; ++m) q[m] = (__bf16)(__expf(p[m] - mx) * rs);
  }
  __syncthreads();

  // ---- Stage 6: attn @ V per head -> sO (aliases sX) ----------------------
  __bf16* sO = sX;
  for (int t = wave; t < 32; t += 4) {
    const int h = t >> 3, rem = t & 7, tm = rem & 3, tn = rem >> 2;  // tn 0..1
    v8f acc = {};
#pragma unroll
    for (int kk = 0; kk < 2; ++kk) {
      v16bf a = frag_a(sAttB + h * 4096, tm * 16, 64, kk * 32, lane);
      v16bf b = frag_b_strided(sV + h * 32, tn * 16, 128, kk * 32, lane);
      acc = __builtin_amdgcn_wmma_f32_16x16x32_bf16(false, a, false, b,
                                                    (short)0, acc, false, false);
    }
    const int col  = h * 32 + tn * 16 + (lane & 15);
    const int half = lane >> 4;
#pragma unroll
    for (int r = 0; r < 8; ++r)
      sO[(tm * 16 + half * 8 + r) * 128 + col] = (__bf16)acc[r];
  }
  __syncthreads();

  // ---- Stage 7: output projection (64x128 @ 128x128) -> sP f32 ------------
  float* sP = sAtt;  // sAtt dead after softmax
  for (int t = wave; t < 32; t += 4) {
    const int tm = t & 3, tn = t >> 2;
    v8f acc = {};
#pragma unroll
    for (int kk = 0; kk < 4; ++kk) {
      v16bf a = frag_a(sO, tm * 16, 128, kk * 32, lane);
      v16bf b = frag_b(wbf + 57344, tn * 16, 128, kk * 32, lane);
      acc = __builtin_amdgcn_wmma_f32_16x16x32_bf16(false, a, false, b,
                                                    (short)0, acc, false, false);
    }
    const int col  = tn * 16 + (lane & 15);
    const int half = lane >> 4;
#pragma unroll
    for (int r = 0; r < 8; ++r)
      sP[(tm * 16 + half * 8 + r) * 128 + col] = acc[r];
  }
  __syncthreads();

  // ---- Stage 8: conv-half + trans-half + biases + residual, scatter -------
  for (int i = tid; i < 4096; i += 128) {
    const int c = i >> 6, n = i & 63;
    const int gd = wd * 4 + (n >> 4);
    const int gh = wh * 4 + ((n >> 2) & 3);
    const int gw = ww * 4 + (n & 3);
    const size_t g = (((size_t)c * 64 + gd) * 64 + gh) * 64 + gw;
    out[g] = sP[n * 128 + c] + sP[n * 128 + 64 + c] +
             proj_b[c] + proj_b[64 + c] + conv_feat[g];
  }
}

// ---------------------------------------------------------------------------
extern "C" void kernel_launch(void* const* d_in, const int* in_sizes, int n_in,
                              void* d_out, int out_size, void* d_ws, size_t ws_size,
                              hipStream_t stream) {
  (void)in_sizes; (void)n_in; (void)out_size; (void)ws_size;
  const float* conv_feat  = (const float*)d_in[0];
  const float* trans_feat = (const float*)d_in[1];
  const float* conv_w     = (const float*)d_in[2];
  const float* conv_b     = (const float*)d_in[3];
  const float* trans_w    = (const float*)d_in[4];
  const float* trans_b    = (const float*)d_in[5];
  const float* qkv_w      = (const float*)d_in[6];
  const float* qkv_b      = (const float*)d_in[7];
  const float* proj_w     = (const float*)d_in[8];
  const float* proj_b     = (const float*)d_in[9];
  const float* rel_table  = (const float*)d_in[10];
  const int*   rel_index  = (const int*)d_in[11];
  unsigned short* wbf = (unsigned short*)d_ws;
  float* out = (float*)d_out;

  const int smem_bytes = 163840;  // 160 KB dynamic LDS (<= 320 KB/WGP)
  (void)hipFuncSetAttribute((const void*)fused_window_attn,
                            hipFuncAttributeMaxDynamicSharedMemorySize, smem_bytes);

  convert_weights<<<(73728 + 255) / 256, 256, 0, stream>>>(conv_w, trans_w,
                                                           qkv_w, proj_w, wbf);
  fused_window_attn<<<4096, 128, smem_bytes, stream>>>(
      conv_feat, trans_feat, conv_b, trans_b, qkv_b, proj_b,
      rel_table, rel_index, wbf, out);
}
